// BriaFiboSingleTransformerBlock_87299505258590
// MI455X (gfx1250) — matrix operand order repaired and checked by reading, not compiled, not run on hardware
//
#include <hip/hip_runtime.h>
#include <math.h>
#include <stdint.h>

typedef unsigned short u16;
typedef __attribute__((ext_vector_type(16))) __bf16 v16bf;
typedef __attribute__((ext_vector_type(8)))  float  v8f;

union Frag16 { v16bf v; uint4 q[2]; };
union Pack16 { u16 u[16]; uint4 q[2]; };

#define DIM 3072
#define SEQ 2048
#define HEADS 24
#define HDIM 128
#define MLPH 12288
#define CATW 15360

// ---- Tensor Data Mover availability (probe-verified builtin; arity differs per toolchain)
#if defined(__has_builtin)
#if __has_builtin(__builtin_amdgcn_tensor_load_to_lds)
#define USE_TDM 1
#endif
#endif
#ifndef USE_TDM
#define USE_TDM 0
#endif
#if __has_include(<hip/amd_detail/amd_gfx1250_TDM.h>)
#define TDM_ARGS6 1
#else
#define TDM_ARGS6 0
#endif

__device__ __forceinline__ u16 f32_to_bf16(float f) {
    unsigned int u = __float_as_uint(f);
    unsigned int r = u + 0x7FFFu + ((u >> 16) & 1u);
    return (u16)(r >> 16);
}
__device__ __forceinline__ v8f zero8() {
    v8f z = {0.f,0.f,0.f,0.f,0.f,0.f,0.f,0.f};
    return z;
}
__device__ __forceinline__ float gelu_tanh(float x) {
    const float c = 0.7978845608028654f;
    float x3 = x * x * x;
    return 0.5f * x * (1.0f + tanhf(c * (x + 0.044715f * x3)));
}

#if USE_TDM
typedef __attribute__((ext_vector_type(4))) unsigned int u32x4;
typedef __attribute__((ext_vector_type(4))) int i32x4;
typedef __attribute__((ext_vector_type(8))) int i32x8;

// 2D TDM load: 32-elem (bf16) rows x 128 rows, row stride lda elems, LDS rows
// padded to 48 elems via D# pad fields (16 DWORDs data -> +8 DWORDs pad).
__device__ __forceinline__ void tdm_load_a_tile(const u16* gsrc, unsigned int lda_elems,
                                                unsigned int lds_byte) {
    unsigned long long ga = (unsigned long long)(uintptr_t)gsrc;
    u32x4 g0;
    g0[0] = 1u;                                   // count=1, user mode
    g0[1] = lds_byte;                             // lds_addr
    g0[2] = (unsigned int)ga;                     // global_addr[31:0]
    g0[3] = (unsigned int)((ga >> 32) & 0x1FFFFFFull) | (2u << 30); // addr hi | type=2
    i32x8 g1;
    g1[0] = (int)((1u << 16) | (1u << 20) | (3u << 22) | (7u << 25)); // 2B elems, pad_en, int=16dw, amt=8dw
    g1[1] = (int)((lda_elems & 0xFFFFu) << 16);   // abar=0 | tensor_dim0[15:0]
    g1[2] = (int)(((lda_elems >> 16) & 0xFFFFu) | (128u << 16)); // tdim0 hi | tensor_dim1 lo
    g1[3] = (int)(32u << 16);                     // tensor_dim1 hi=0 | tile_dim0=32
    g1[4] = (int)128u;                            // tile_dim1=128 | tile_dim2=0
    g1[5] = (int)lda_elems;                       // tensor_dim0_stride lo
    g1[6] = 0;                                    // stride hi | dim1_stride lo
    g1[7] = 0;
    i32x4 z4 = {0, 0, 0, 0};
#if TDM_ARGS6
    i32x8 z8 = {0,0,0,0,0,0,0,0};
    __builtin_amdgcn_tensor_load_to_lds(g0, g1, z4, z4, z8, 0);
#else
    __builtin_amdgcn_tensor_load_to_lds(g0, g1, z4, z4, 0);
#endif
}
#endif

// ---------------- 1: silu(temb) ----------------
__global__ __launch_bounds__(256) void silu_kernel(const float* __restrict__ temb,
                                                   float* __restrict__ st) {
    int i = blockIdx.x * 256 + threadIdx.x;
    if (i < DIM) {
        float x = temb[i];
        st[i] = x / (1.0f + __expf(-x));
    }
}

// ---------------- 2: emb = silu(temb) @ norm_w.T + norm_b ----------------
__global__ __launch_bounds__(256) void emb_gemv_kernel(const float* __restrict__ st,
                                                       const float* __restrict__ w,
                                                       const float* __restrict__ b,
                                                       float* __restrict__ emb) {
    int j = blockIdx.x * 8 + (threadIdx.x >> 5);
    int lane = threadIdx.x & 31;
    const float* wr = w + (size_t)j * DIM;
    float acc = 0.f;
    for (int i = lane; i < DIM; i += 32) acc += st[i] * wr[i];
    #pragma unroll
    for (int m = 1; m < 32; m <<= 1) acc += __shfl_xor(acc, m, 32);
    if (lane == 0) emb[j] = acc + b[j];
}

// ---------------- 3: LayerNorm + modulate -> nh (bf16) ----------------
__global__ __launch_bounds__(256) void ln_mod_kernel(const float* __restrict__ x,
                                                     const float* __restrict__ emb,
                                                     u16* __restrict__ nh) {
    int s = blockIdx.x;
    int t = threadIdx.x;
    const float* row = x + (size_t)s * DIM;
    float vals[12];
    float sum = 0.f, ssq = 0.f;
    #pragma unroll
    for (int i = 0; i < 12; ++i) {
        float v = row[t + i * 256];
        vals[i] = v; sum += v; ssq += v * v;
    }
    __shared__ float redS[8], redQ[8];
    #pragma unroll
    for (int m = 1; m < 32; m <<= 1) { sum += __shfl_xor(sum, m, 32); ssq += __shfl_xor(ssq, m, 32); }
    if ((t & 31) == 0) { redS[t >> 5] = sum; redQ[t >> 5] = ssq; }
    __syncthreads();
    float sAll = 0.f, qAll = 0.f;
    #pragma unroll
    for (int i = 0; i < 8; ++i) { sAll += redS[i]; qAll += redQ[i]; }
    float mean = sAll * (1.0f / DIM);
    float var  = qAll * (1.0f / DIM) - mean * mean;
    float inv  = rsqrtf(var + 1e-6f);
    #pragma unroll
    for (int i = 0; i < 12; ++i) {
        int n = t + i * 256;
        float v = (vals[i] - mean) * inv;
        v = v * (1.0f + emb[DIM + n]) + emb[n];
        nh[(size_t)s * DIM + n] = f32_to_bf16(v);
    }
}

// -------- generic bf16 WMMA GEMM, double-buffered:  C[M,N] = A[M,K] * W[N,K]^T
// MODE 0: f32 head-major (q/k)  MODE 1: bf16 head-major (v)
// MODE 2: gelu->bf16 into cat[:,3072+n]  MODE 3: gate*(acc+b)+resid -> f32 d_out
template <int MODE>
__global__ __launch_bounds__(256) void gemm_bf16_kernel(
    const u16* __restrict__ A, int lda,
    const float* __restrict__ W,
    const float* __restrict__ bias,
    void* __restrict__ out,
    const float* __restrict__ gate,
    const float* __restrict__ resid,
    int K)
{
    __shared__ u16 ldsA[2][128 * 48];
    __shared__ u16 ldsB[2][128 * 48];
    const int bm = blockIdx.y * 128, bn = blockIdx.x * 128;
    const int t = threadIdx.x;
    const int w = t >> 5, lane = t & 31, lm = lane & 15, hl = lane >> 4;
    const int wm = w >> 2, wn = w & 3;
    const int rowT = t >> 1, segT = (t & 1) * 16;

    v8f acc[4][2];
    #pragma unroll
    for (int mi = 0; mi < 4; ++mi)
        #pragma unroll
        for (int ni = 0; ni < 2; ++ni) acc[mi][ni] = zero8();

    // ---- prologue: stage tile 0
    float4 regB[4];
    {
        const float4* gw = (const float4*)(W + (size_t)(bn + rowT) * K + segT);
        regB[0] = gw[0]; regB[1] = gw[1]; regB[2] = gw[2]; regB[3] = gw[3];
    }
#if USE_TDM
    if (w == 0)
        tdm_load_a_tile(A + (size_t)bm * lda, (unsigned int)lda,
                        (unsigned int)(uintptr_t)&ldsA[0][0]);
#else
    uint4 regA[2];
    {
        const uint4* ga = (const uint4*)(A + (size_t)(bm + rowT) * lda + segT);
        regA[0] = ga[0]; regA[1] = ga[1];
    }
#endif

    int it = 0;
    for (int k0 = 0; k0 < K; k0 += 32, ++it) {
        const int cur = it & 1, nxt = cur ^ 1;
        // stage B_i (f32->bf16) into LDS[cur]
        {
            Pack16 pk;
            #pragma unroll
            for (int i = 0; i < 4; ++i) {
                float4 f = regB[i];
                pk.u[i*4+0] = f32_to_bf16(f.x); pk.u[i*4+1] = f32_to_bf16(f.y);
                pk.u[i*4+2] = f32_to_bf16(f.z); pk.u[i*4+3] = f32_to_bf16(f.w);
            }
            *(uint4*)&ldsB[cur][rowT * 48 + segT]     = pk.q[0];
            *(uint4*)&ldsB[cur][rowT * 48 + segT + 8] = pk.q[1];
        }
#if !USE_TDM
        *(uint4*)&ldsA[cur][rowT * 48 + segT]     = regA[0];
        *(uint4*)&ldsA[cur][rowT * 48 + segT + 8] = regA[1];
#endif
        // issue tile i+1 (overlaps with compute of tile i)
        if (k0 + 32 < K) {
            const float4* gw = (const float4*)(W + (size_t)(bn + rowT) * K + (k0 + 32) + segT);
            regB[0] = gw[0]; regB[1] = gw[1]; regB[2] = gw[2]; regB[3] = gw[3];
#if USE_TDM
            if (w == 0)
                tdm_load_a_tile(A + (size_t)bm * lda + (k0 + 32), (unsigned int)lda,
                                (unsigned int)(uintptr_t)&ldsA[nxt][0]);
#else
            const uint4* ga = (const uint4*)(A + (size_t)(bm + rowT) * lda + (k0 + 32) + segT);
            regA[0] = ga[0]; regA[1] = ga[1];
#endif
        }
#if USE_TDM
        if (w == 0) {
            if (k0 + 32 < K) __builtin_amdgcn_s_wait_tensorcnt((short)1);
            else             __builtin_amdgcn_s_wait_tensorcnt((short)0);
        }
#endif
        __syncthreads();   // publish A_i (TDM/regs) + B_i

        Frag16 af[4];
        #pragma unroll
        for (int mi = 0; mi < 4; ++mi) {
            int r = wm * 64 + mi * 16 + lm;
            af[mi].q[0] = *(const uint4*)&ldsA[cur][r * 48 + hl * 8];
            af[mi].q[1] = *(const uint4*)&ldsA[cur][r * 48 + 16 + hl * 8];
        }
        v16bf bfv[2];
        #pragma unroll
        for (int ni = 0; ni < 2; ++ni) {
            int c = wn * 32 + ni * 16 + lm;
            Frag16 bt;
            bt.q[0] = *(const uint4*)&ldsB[cur][c * 48 + hl * 8];
            bt.q[1] = *(const uint4*)&ldsB[cur][c * 48 + 16 + hl * 8];
            bfv[ni] = bt.v;
        }
        #pragma unroll
        for (int mi = 0; mi < 4; ++mi)
            #pragma unroll
            for (int ni = 0; ni < 2; ++ni)
                acc[mi][ni] = __builtin_amdgcn_wmma_f32_16x16x32_bf16(
                    false, af[mi].v, false, bfv[ni], (short)0, acc[mi][ni], false, false);

        __syncthreads();   // done reading buffers[cur] before it is refilled
    }

    // epilogue
    #pragma unroll
    for (int mi = 0; mi < 4; ++mi) {
        #pragma unroll
        for (int ni = 0; ni < 2; ++ni) {
            #pragma unroll
            for (int r = 0; r < 8; ++r) {
                int m = bm + wm * 64 + mi * 16 + hl * 8 + r;
                int n = bn + wn * 32 + ni * 16 + lm;
                float v = acc[mi][ni][r] + bias[n];
                if (MODE == 0) {
                    ((float*)out)[((size_t)(n >> 7) * SEQ + m) * HDIM + (n & 127)] = v;
                } else if (MODE == 1) {
                    ((u16*)out)[((size_t)(n >> 7) * SEQ + m) * HDIM + (n & 127)] = f32_to_bf16(v);
                } else if (MODE == 2) {
                    ((u16*)out)[(size_t)m * CATW + DIM + n] = f32_to_bf16(gelu_tanh(v));
                } else {
                    float g = gate[n];
                    float rs = resid[(size_t)m * DIM + n];
                    ((float*)out)[(size_t)m * DIM + n] = g * v + rs;
                }
            }
        }
    }
}

// ---------------- RMS-norm + RoPE, f32 head-major -> bf16 head-major ----------
__global__ __launch_bounds__(128) void rmsrope_kernel(const float* __restrict__ xf,
                                                      const float* __restrict__ rw,
                                                      const float* __restrict__ cosb,
                                                      const float* __restrict__ sinb,
                                                      u16* __restrict__ outh) {
    int h = blockIdx.y, s = blockIdx.x, d = threadIdx.x;
    const float* x = xf + ((size_t)h * SEQ + s) * HDIM;
    float v = x[d];
    float sq = v * v;
    __shared__ float red[4];
    #pragma unroll
    for (int m = 1; m < 32; m <<= 1) sq += __shfl_xor(sq, m, 32);
    if ((threadIdx.x & 31) == 0) red[threadIdx.x >> 5] = sq;
    __syncthreads();
    float tot = red[0] + red[1] + red[2] + red[3];
    float r = rsqrtf(tot * (1.0f / HDIM) + 1e-6f);
    float xn = v * r * rw[d];
    int dn = d ^ 1;
    float nb = x[dn] * r * rw[dn];
    float rot = ((d & 1) == 0) ? -nb : nb;
    float o = xn * cosb[s * HDIM + d] + rot * sinb[s * HDIM + d];
    outh[((size_t)h * SEQ + s) * HDIM + d] = f32_to_bf16(o);
}

// ---------------- flash attention (per head, 64 q-rows per WG, 32-key blocks) --
__global__ __launch_bounds__(128) void flash_attn_kernel(const u16* __restrict__ qh,
                                                         const u16* __restrict__ kh,
                                                         const u16* __restrict__ vh,
                                                         u16* __restrict__ cat) {
    __shared__ u16 ldsVt[128 * 48];      // V^T: [dim][key], stride 48
    __shared__ u16 ldsP[4 * 16 * 48];    // per-wave P tiles
    const int h = blockIdx.y;
    const int q0 = blockIdx.x * 64;
    const int t = threadIdx.x, w = t >> 5, lane = t & 31, lm = lane & 15, hl = lane >> 4;
    const u16* Q  = qh + (size_t)h * SEQ * HDIM;
    const u16* Kp = kh + (size_t)h * SEQ * HDIM;
    const u16* Vp = vh + (size_t)h * SEQ * HDIM;
    const float scale = 0.08838834764831845f; // 1/sqrt(128)

    Frag16 qf[4];
    {
        int qrow = q0 + w * 16 + lm;
        #pragma unroll
        for (int kc = 0; kc < 4; ++kc) {
            qf[kc].q[0] = *(const uint4*)&Q[(size_t)qrow * HDIM + kc * 32 + hl * 8];
            qf[kc].q[1] = *(const uint4*)&Q[(size_t)qrow * HDIM + kc * 32 + 16 + hl * 8];
        }
    }

    v8f o[8];
    #pragma unroll
    for (int i = 0; i < 8; ++i) o[i] = zero8();
    float mrow[8], lrow[8];
    #pragma unroll
    for (int r = 0; r < 8; ++r) { mrow[r] = -1e30f; lrow[r] = 0.f; }

    const int kg = (t & 7) * 4;      // 4-key group for V transpose
    const int dg = (t >> 3) * 8;     // 8-dim group

    for (int kb = 0; kb < SEQ; kb += 32) {
        __syncthreads();
        // V block -> LDS transposed: 4x8 register transpose, b64 stores
        {
            union { uint4 q; unsigned int u[4]; } a0, a1, a2, a3;
            a0.q = *(const uint4*)&Vp[(size_t)(kb + kg + 0) * HDIM + dg];
            a1.q = *(const uint4*)&Vp[(size_t)(kb + kg + 1) * HDIM + dg];
            a2.q = *(const uint4*)&Vp[(size_t)(kb + kg + 2) * HDIM + dg];
            a3.q = *(const uint4*)&Vp[(size_t)(kb + kg + 3) * HDIM + dg];
            #pragma unroll
            for (int dd = 0; dd < 8; ++dd) {
                int wi = dd >> 1, sh = (dd & 1) * 16;
                unsigned int x0 = (a0.u[wi] >> sh) & 0xFFFFu;
                unsigned int x1 = (a1.u[wi] >> sh) & 0xFFFFu;
                unsigned int x2 = (a2.u[wi] >> sh) & 0xFFFFu;
                unsigned int x3 = (a3.u[wi] >> sh) & 0xFFFFu;
                *(uint2*)&ldsVt[(dg + dd) * 48 + kg] = make_uint2(x0 | (x1 << 16), x2 | (x3 << 16));
            }
        }
        // S = Q K^T for this wave's 16x32 strip (K frags straight from global/L2)
        v8f sacc[2];
        sacc[0] = zero8(); sacc[1] = zero8();
        #pragma unroll
        for (int nt = 0; nt < 2; ++nt) {
            int krow = kb + nt * 16 + lm;
            #pragma unroll
            for (int kc = 0; kc < 4; ++kc) {
                Frag16 kt;
                kt.q[0] = *(const uint4*)&Kp[(size_t)krow * HDIM + kc * 32 + hl * 8];
                kt.q[1] = *(const uint4*)&Kp[(size_t)krow * HDIM + kc * 32 + 16 + hl * 8];
                v16bf kf = kt.v;
                sacc[nt] = __builtin_amdgcn_wmma_f32_16x16x32_bf16(
                    false, qf[kc].v, false, kf, (short)0, sacc[nt], false, false);
            }
        }
        // online softmax
        float rm[8];
        #pragma unroll
        for (int r = 0; r < 8; ++r)
            rm[r] = fmaxf(sacc[0][r], sacc[1][r]) * scale;
        #pragma unroll
        for (int m = 1; m < 16; m <<= 1)
            #pragma unroll
            for (int r = 0; r < 8; ++r)
                rm[r] = fmaxf(rm[r], __shfl_xor(rm[r], m, 32));
        float rs[8], alpha[8];
        #pragma unroll
        for (int r = 0; r < 8; ++r) {
            float mnew = fmaxf(mrow[r], rm[r]);
            alpha[r] = __expf(mrow[r] - mnew);
            float e0 = __expf(sacc[0][r] * scale - mnew);
            float e1 = __expf(sacc[1][r] * scale - mnew);
            rs[r] = e0 + e1;
            mrow[r] = mnew;
            ldsP[w * 768 + (hl * 8 + r) * 48 + lm]      = f32_to_bf16(e0);
            ldsP[w * 768 + (hl * 8 + r) * 48 + 16 + lm] = f32_to_bf16(e1);
        }
        #pragma unroll
        for (int m = 1; m < 16; m <<= 1)
            #pragma unroll
            for (int r = 0; r < 8; ++r)
                rs[r] += __shfl_xor(rs[r], m, 32);
        #pragma unroll
        for (int r = 0; r < 8; ++r) lrow[r] = lrow[r] * alpha[r] + rs[r];
        #pragma unroll
        for (int dt = 0; dt < 8; ++dt)
            #pragma unroll
            for (int r = 0; r < 8; ++r) o[dt][r] *= alpha[r];

        Frag16 pf;
        pf.q[0] = *(const uint4*)&ldsP[w * 768 + lm * 48 + hl * 8];
        pf.q[1] = *(const uint4*)&ldsP[w * 768 + lm * 48 + 16 + hl * 8];
        __syncthreads(); // Vt ready
        #pragma unroll
        for (int dt = 0; dt < 8; ++dt) {
            Frag16 vt;
            vt.q[0] = *(const uint4*)&ldsVt[(dt * 16 + lm) * 48 + hl * 8];
            vt.q[1] = *(const uint4*)&ldsVt[(dt * 16 + lm) * 48 + 16 + hl * 8];
            v16bf vf = vt.v;
            o[dt] = __builtin_amdgcn_wmma_f32_16x16x32_bf16(
                false, pf.v, false, vf, (short)0, o[dt], false, false);
        }
    }
    // write attn into cat[:, :3072] as bf16
    #pragma unroll
    for (int r = 0; r < 8; ++r) {
        float inv = 1.0f / lrow[r];
        int m = q0 + w * 16 + hl * 8 + r;
        #pragma unroll
        for (int dt = 0; dt < 8; ++dt) {
            int n = h * HDIM + dt * 16 + lm;
            cat[(size_t)m * CATW + n] = f32_to_bf16(o[dt][r] * inv);
        }
    }
}

// ------------------------------- launcher -------------------------------------
extern "C" void kernel_launch(void* const* d_in, const int* in_sizes, int n_in,
                              void* d_out, int out_size, void* d_ws, size_t ws_size,
                              hipStream_t stream) {
    const float* hidden   = (const float*)d_in[0];
    const float* temb     = (const float*)d_in[1];
    const float* rope_cos = (const float*)d_in[2];
    const float* rope_sin = (const float*)d_in[3];
    const float* norm_w   = (const float*)d_in[4];
    const float* norm_b   = (const float*)d_in[5];
    const float* mlp_w    = (const float*)d_in[6];
    const float* mlp_b    = (const float*)d_in[7];
    const float* q_w      = (const float*)d_in[8];
    const float* q_b      = (const float*)d_in[9];
    const float* k_w      = (const float*)d_in[10];
    const float* k_b      = (const float*)d_in[11];
    const float* v_w      = (const float*)d_in[12];
    const float* v_b      = (const float*)d_in[13];
    const float* rms_q_w  = (const float*)d_in[14];
    const float* rms_k_w  = (const float*)d_in[15];
    const float* out_w    = (const float*)d_in[16];
    const float* out_b    = (const float*)d_in[17];

    char* ws = (char*)d_ws;
    float* st   = (float*)(ws + 0);                    // 12288 B
    float* emb  = (float*)(ws + 12288);                // 36864 B
    u16*   nh   = (u16*)  (ws + 49152);                // 2048*3072 bf16
    u16*   cat  = (u16*)  (ws + 12632064);             // 2048*15360 bf16
    float* qf   = (float*)(ws + 75546624);             // 24*2048*128 f32
    float* kf   = (float*)(ws + 100712448);            // 24*2048*128 f32
    u16*   qh   = (u16*)  (ws + 125878272);            // bf16 head-major
    u16*   kh   = (u16*)  (ws + 138461184);
    u16*   vh   = (u16*)  (ws + 151044096);
    (void)in_sizes; (void)n_in; (void)out_size; (void)ws_size;

    silu_kernel<<<dim3(12), dim3(256), 0, stream>>>(temb, st);
    emb_gemv_kernel<<<dim3(1152), dim3(256), 0, stream>>>(st, norm_w, norm_b, emb);
    ln_mod_kernel<<<dim3(SEQ), dim3(256), 0, stream>>>(hidden, emb, nh);

    gemm_bf16_kernel<0><<<dim3(DIM/128, SEQ/128), dim3(256), 0, stream>>>(
        nh, DIM, q_w, q_b, qf, nullptr, nullptr, DIM);
    gemm_bf16_kernel<0><<<dim3(DIM/128, SEQ/128), dim3(256), 0, stream>>>(
        nh, DIM, k_w, k_b, kf, nullptr, nullptr, DIM);
    gemm_bf16_kernel<1><<<dim3(DIM/128, SEQ/128), dim3(256), 0, stream>>>(
        nh, DIM, v_w, v_b, vh, nullptr, nullptr, DIM);
    gemm_bf16_kernel<2><<<dim3(MLPH/128, SEQ/128), dim3(256), 0, stream>>>(
        nh, DIM, mlp_w, mlp_b, cat, nullptr, nullptr, DIM);

    rmsrope_kernel<<<dim3(SEQ, HEADS), dim3(128), 0, stream>>>(qf, rms_q_w, rope_cos, rope_sin, qh);
    rmsrope_kernel<<<dim3(SEQ, HEADS), dim3(128), 0, stream>>>(kf, rms_k_w, rope_cos, rope_sin, kh);

    flash_attn_kernel<<<dim3(SEQ/64, HEADS), dim3(128), 0, stream>>>(qh, kh, vh, cat);

    gemm_bf16_kernel<3><<<dim3(DIM/128, SEQ/128), dim3(256), 0, stream>>>(
        cat, CATW, out_w, out_b, d_out, emb + 2*DIM, hidden, CATW);
}
